// HeteroGraphODENetwork_55817394979279
// MI455X (gfx1250) — matrix operand order repaired and checked by reading, not compiled, not run on hardware
//
#include <hip/hip_runtime.h>
#include <math.h>

typedef __attribute__((ext_vector_type(16))) _Float16 v16h;
typedef __attribute__((ext_vector_type(8)))  float    v8f;

#define DEVINL __device__ __forceinline__

constexpr int NA  = 100000;
constexpr int NPK = 50000;
constexpr int NL  = 200000;
constexpr int NE  = 1600000;
constexpr int H   = 64;
constexpr int AH  = 16;               // action head width
constexpr int NTOT = NA + NPK + NL;   // 350000, divisible by 16

constexpr int RK_WAVES = 4;           // waves per block in RK4 kernel
constexpr int LDSW     = 68;          // padded LDS row stride (floats); 68*4=272 keeps 16B row alignment
constexpr int SG_WAVES = 8;
constexpr int HD_WAVES = 8;

// ---------------------------------------------------------------------------
// Fast tanh: gfx1250 has a hardware V_TANH_F32 transcendental.
// ---------------------------------------------------------------------------
DEVINL float fast_tanh(float x) {
#if __has_builtin(__builtin_amdgcn_tanhf)
  return __builtin_amdgcn_tanhf(x);
#else
  return tanhf(x);
#endif
}

// ---------------------------------------------------------------------------
// WMMA helpers: D(16x16 f32) = A(16x32 f16) * B(32x16 f16) + C
// ---------------------------------------------------------------------------
DEVINL v8f wmma16(v16h a, v16h b, v8f c) {
  return __builtin_amdgcn_wmma_f32_16x16x32_f16(
      /*neg_a=*/false, a, /*neg_b=*/false, b,
      /*c_mod=*/(short)0, c, /*reuse_a=*/false, /*reuse_b=*/false);
}

// Build an A fragment for wave32 f16 16x32 layout.
// Lane L owns matrix row m = L&15; hi = L>>4 selects the K-subrange.
// VGPR v holds halves j=2v,2v+1 with K = (v/4)*16 + hi*8 + (v%4)*2 (+parity).
// rowp points at the f32 row (LDS or global); scale is folded into the convert.
DEVINL v16h afrag(const float* rowp, int kofs, int hi, float scale) {
  v16h a;
#pragma unroll
  for (int v = 0; v < 8; ++v) {
    int k = kofs + ((v >> 2) << 4) + (hi << 3) + ((v & 3) << 1);
    a[2 * v]     = (_Float16)(rowp[k] * scale);
    a[2 * v + 1] = (_Float16)(rowp[k + 1] * scale);
  }
  return a;
}

// B fragments are pre-packed so each lane reads 16 contiguous halves (32B).
DEVINL v16h bfrag(const _Float16* pk, int frag, int lane) {
  return *(const v16h*)(pk + ((size_t)frag * 32 + lane) * 16);
}

// ---------------------------------------------------------------------------
// Weight packing: convert f32 W[K,N] to f16 in native B-fragment layout.
// Fragment f = nt*KT + kt; within a fragment, lane L holds column
// n = nt*16 + (L&15) and halves j -> k = kt*32 + (j>>3)*16 + (L>>4)*8 + (j&7).
// ---------------------------------------------------------------------------
__global__ void pack_b(const float* __restrict__ W, _Float16* __restrict__ out,
                       int Kdim, int Ndim, int nmat) {
  int per = Kdim * Ndim;
  int KT  = Kdim >> 5;
  int total = per * nmat;
  for (int i = blockIdx.x * blockDim.x + threadIdx.x; i < total;
       i += gridDim.x * blockDim.x) {
    int mat  = i / per;
    int ii   = i - mat * per;
    int j    = ii & 15;
    int lane = (ii >> 4) & 31;
    int frag = ii >> 9;
    int kt = frag % KT, nt = frag / KT;
    int k = kt * 32 + ((j >> 3) << 4) + ((lane >> 4) << 3) + (j & 7);
    int n = nt * 16 + (lane & 15);
    out[i] = (_Float16)W[(size_t)mat * per + (size_t)k * Ndim + n];
  }
}

// ---------------------------------------------------------------------------
// Node embedding: h[N,64] = x[N,Fin] @ W[Fin,64] + b   (Fin in {7,4,2})
// ---------------------------------------------------------------------------
__global__ void embed(const float* __restrict__ x, const float* __restrict__ W,
                      const float* __restrict__ b, float* __restrict__ h,
                      int N, int Fin) {
  int total = N * H;
  for (int i = blockIdx.x * blockDim.x + threadIdx.x; i < total;
       i += gridDim.x * blockDim.x) {
    int node = i >> 6, c = i & 63;
    float acc = b[c];
    for (int f = 0; f < Fin; ++f)
      acc += x[(size_t)node * Fin + f] * W[f * H + c];
    h[i] = acc;
  }
}

// ---------------------------------------------------------------------------
// Edge scatter: one wave per edge; lane handles 2 features (float2).
// sum[dst] += h[src]; cnt[dst] += 1. Prefetch next gather row to hide the
// random-access latency behind the atomic traffic.
// ---------------------------------------------------------------------------
__global__ void edge_scatter(const float* __restrict__ hsrc,
                             const int* __restrict__ src,
                             const int* __restrict__ dst,
                             float* __restrict__ sum, float* __restrict__ cnt,
                             int ne) {
  int gtid = blockIdx.x * blockDim.x + threadIdx.x;
  int lane = gtid & 31;
  int nw   = (gridDim.x * blockDim.x) >> 5;
  for (int e = gtid >> 5; e < ne; e += nw) {
    int s = src[e], d = dst[e];
    int e2 = e + nw;
    if (e2 < ne)
      __builtin_prefetch(hsrc + (size_t)src[e2] * H + lane * 2, 0, 0);
    float2 v = ((const float2*)(hsrc + (size_t)s * H))[lane];
    float* dp = sum + (size_t)d * H + lane * 2;
    atomicAdd(dp, v.x);
    atomicAdd(dp + 1, v.y);
    if (lane == 0) atomicAdd(cnt + d, 1.0f);
  }
}

// ---------------------------------------------------------------------------
// SAGE dense transform, one wave per 16 rows:
//   comb += (sum/max(cnt,1)) @ Wl + bl + h_dst @ Wr      (64x64 weights)
// All 16 B-fragments are hoisted into registers so the 16 WMMAs chain
// without per-op loadcnt stalls.
// ---------------------------------------------------------------------------
__global__ void __launch_bounds__(SG_WAVES * 32)
sage_gemm(const float* __restrict__ sum, const float* __restrict__ cnt,
          const float* __restrict__ hdst, const _Float16* __restrict__ wl,
          const float* __restrict__ blv, const _Float16* __restrict__ wr,
          float* __restrict__ comb, int nrb) {
  int wave = threadIdx.x >> 5, lane = threadIdx.x & 31;
  int rb = blockIdx.x * SG_WAVES + wave;
  if (rb >= nrb) return;
  int m = lane & 15, hi = lane >> 4;

  v16h wfl[8], wfr[8];
#pragma unroll
  for (int f = 0; f < 8; ++f) {
    wfl[f] = bfrag(wl, f, lane);
    wfr[f] = bfrag(wr, f, lane);
  }

  size_t row = (size_t)rb * 16 + m;
  float ic = 1.0f / fmaxf(cnt[row], 1.0f);
  const float* srow = sum + row * H;
  const float* drow = hdst + row * H;
  v16h am0 = afrag(srow, 0, hi, ic);     // mean operand (scaled)
  v16h am1 = afrag(srow, 32, hi, ic);
  v16h ad0 = afrag(drow, 0, hi, 1.0f);   // self operand
  v16h ad1 = afrag(drow, 32, hi, 1.0f);
#pragma unroll
  for (int nt = 0; nt < 4; ++nt) {
    v8f acc;
    float bv = blv[nt * 16 + m];
#pragma unroll
    for (int r = 0; r < 8; ++r) acc[r] = bv;
    acc = wmma16(am0, wfl[nt * 2 + 0], acc);
    acc = wmma16(am1, wfl[nt * 2 + 1], acc);
    acc = wmma16(ad0, wfr[nt * 2 + 0], acc);
    acc = wmma16(ad1, wfr[nt * 2 + 1], acc);
#pragma unroll
    for (int r = 0; r < 8; ++r) {
      size_t orow = (size_t)rb * 16 + hi * 8 + r;
      comb[orow * H + nt * 16 + m] += acc[r];
    }
  }
}

// h = relu(comb * scale)
__global__ void relu_scale(const float* __restrict__ c, float* __restrict__ h,
                           float s, int total) {
  for (int i = blockIdx.x * blockDim.x + threadIdx.x; i < total;
       i += gridDim.x * blockDim.x)
    h[i] = fmaxf(c[i] * s, 0.0f);
}

// ---------------------------------------------------------------------------
// ode_f on one 16-row tile held in LDS (sb, row stride LDSW):
//   t1 = tanh(x @ W1[64x32] + b1); t2 = tanh(t1 @ W2[32x32] + b2);
//   f  = t2 @ W3[32x64] + b3   (returned in C-layout registers f[0..3])
// Weight fragments + per-lane bias values come preloaded in registers.
// Intermediates are staged in-place in sb cols 0..31 (wave-synchronous LDS).
// ---------------------------------------------------------------------------
DEVINL void ode_eval(float* sb, const v16h* wf1, const v16h* wf2,
                     const v16h* wf3, const float* b1v, const float* b2v,
                     const float* b3v, int lane, v8f* f) {
  const int m = lane & 15, hi = lane >> 4;
  const float* arow = sb + m * LDSW;
  v16h a0 = afrag(arow, 0, hi, 1.0f);
  v16h a1 = afrag(arow, 32, hi, 1.0f);
  float t[2][8];
#pragma unroll
  for (int nt = 0; nt < 2; ++nt) {
    v8f acc;
#pragma unroll
    for (int r = 0; r < 8; ++r) acc[r] = b1v[nt];
    acc = wmma16(a0, wf1[nt * 2 + 0], acc);
    acc = wmma16(a1, wf1[nt * 2 + 1], acc);
#pragma unroll
    for (int r = 0; r < 8; ++r) t[nt][r] = fast_tanh(acc[r]);
  }
#pragma unroll
  for (int nt = 0; nt < 2; ++nt)
#pragma unroll
    for (int r = 0; r < 8; ++r)
      sb[(hi * 8 + r) * LDSW + nt * 16 + m] = t[nt][r];

  v16h a2 = afrag(arow, 0, hi, 1.0f);   // K = 0..31 over t1
#pragma unroll
  for (int nt = 0; nt < 2; ++nt) {
    v8f acc;
#pragma unroll
    for (int r = 0; r < 8; ++r) acc[r] = b2v[nt];
    acc = wmma16(a2, wf2[nt], acc);
#pragma unroll
    for (int r = 0; r < 8; ++r) t[nt][r] = fast_tanh(acc[r]);
  }
#pragma unroll
  for (int nt = 0; nt < 2; ++nt)
#pragma unroll
    for (int r = 0; r < 8; ++r)
      sb[(hi * 8 + r) * LDSW + nt * 16 + m] = t[nt][r];

  v16h a3 = afrag(arow, 0, hi, 1.0f);   // K = 0..31 over t2
#pragma unroll
  for (int nt = 0; nt < 4; ++nt) {
    v8f acc;
#pragma unroll
    for (int r = 0; r < 8; ++r) acc[r] = b3v[nt];
    f[nt] = wmma16(a3, wf3[nt], acc);
  }
}

// ---------------------------------------------------------------------------
// One full RK4 step, fused per wave (16 rows). All four ode_f evals stay
// on-chip; X is read and written exactly once per step (in-place).
// ---------------------------------------------------------------------------
__global__ void __launch_bounds__(RK_WAVES * 32)
rk4_step(float* __restrict__ X, const _Float16* __restrict__ w1,
         const float* __restrict__ b1, const _Float16* __restrict__ w2,
         const float* __restrict__ b2, const _Float16* __restrict__ w3,
         const float* __restrict__ b3, int nrb, float dt) {
  __shared__ float smem[RK_WAVES][2][16 * LDSW];
  int wave = threadIdx.x >> 5, lane = threadIdx.x & 31;
  int rb = blockIdx.x * RK_WAVES + wave;
  if (rb >= nrb) return;
  float* xb = smem[wave][0];
  float* sb = smem[wave][1];
  float* Xr = X + (size_t)rb * 16 * H;
  int m = lane & 15, hi = lane >> 4;

  // Preload all ODE weight fragments (10 x v16h) and per-lane biases once;
  // they are reused by all 4 evals (40 WMMAs) of this wave.
  v16h wf1[4], wf2[2], wf3[4];
#pragma unroll
  for (int f = 0; f < 4; ++f) wf1[f] = bfrag(w1, f, lane);
#pragma unroll
  for (int f = 0; f < 2; ++f) wf2[f] = bfrag(w2, f, lane);
#pragma unroll
  for (int f = 0; f < 4; ++f) wf3[f] = bfrag(w3, f, lane);
  float b1v[2] = {b1[m], b1[16 + m]};
  float b2v[2] = {b2[m], b2[16 + m]};
  float b3v[4] = {b3[m], b3[16 + m], b3[32 + m], b3[48 + m]};

  // Stage the 16x64 tile via float4 (rows stay 16B aligned with LDSW=68).
#pragma unroll
  for (int t4 = 0; t4 < 8; ++t4) {
    int fi = t4 * 32 + lane;          // float4 index within tile
    float4 v = ((const float4*)Xr)[fi];
    int r = fi >> 4, c = (fi & 15) * 4;
    *(float4*)(xb + r * LDSW + c) = v;
    *(float4*)(sb + r * LDSW + c) = v;
  }

  v8f k[4], ks[4];

  ode_eval(sb, wf1, wf2, wf3, b1v, b2v, b3v, lane, k);           // k1
#pragma unroll
  for (int nt = 0; nt < 4; ++nt)
#pragma unroll
    for (int r = 0; r < 8; ++r) {
      ks[nt][r] = k[nt][r];
      int rr = hi * 8 + r, cc = nt * 16 + m;
      sb[rr * LDSW + cc] = xb[rr * LDSW + cc] + 0.5f * dt * k[nt][r];
    }

  ode_eval(sb, wf1, wf2, wf3, b1v, b2v, b3v, lane, k);           // k2
#pragma unroll
  for (int nt = 0; nt < 4; ++nt)
#pragma unroll
    for (int r = 0; r < 8; ++r) {
      ks[nt][r] += 2.0f * k[nt][r];
      int rr = hi * 8 + r, cc = nt * 16 + m;
      sb[rr * LDSW + cc] = xb[rr * LDSW + cc] + 0.5f * dt * k[nt][r];
    }

  ode_eval(sb, wf1, wf2, wf3, b1v, b2v, b3v, lane, k);           // k3
#pragma unroll
  for (int nt = 0; nt < 4; ++nt)
#pragma unroll
    for (int r = 0; r < 8; ++r) {
      ks[nt][r] += 2.0f * k[nt][r];
      int rr = hi * 8 + r, cc = nt * 16 + m;
      sb[rr * LDSW + cc] = xb[rr * LDSW + cc] + dt * k[nt][r];
    }

  ode_eval(sb, wf1, wf2, wf3, b1v, b2v, b3v, lane, k);           // k4
  float s6 = dt * (1.0f / 6.0f);
#pragma unroll
  for (int nt = 0; nt < 4; ++nt)
#pragma unroll
    for (int r = 0; r < 8; ++r) {
      ks[nt][r] += k[nt][r];
      int rr = hi * 8 + r, cc = nt * 16 + m;
      Xr[rr * H + cc] = xb[rr * LDSW + cc] + s6 * ks[nt][r];
    }
}

// ---------------------------------------------------------------------------
// Head MLP: q = relu(emb @ W1[64x32] + b1) @ W2[32x16] + b2
// ---------------------------------------------------------------------------
__global__ void __launch_bounds__(HD_WAVES * 32)
head_mlp(const float* __restrict__ emb, const _Float16* __restrict__ w1,
         const float* __restrict__ b1, const _Float16* __restrict__ w2,
         const float* __restrict__ b2, float* __restrict__ q, int nrb) {
  __shared__ float tbuf[HD_WAVES][16 * 36];
  int wave = threadIdx.x >> 5, lane = threadIdx.x & 31;
  int rb = blockIdx.x * HD_WAVES + wave;
  if (rb >= nrb) return;
  int m = lane & 15, hi = lane >> 4;

  v16h wf1[4];
#pragma unroll
  for (int f = 0; f < 4; ++f) wf1[f] = bfrag(w1, f, lane);
  v16h wf2 = bfrag(w2, 0, lane);

  const float* erow = emb + ((size_t)rb * 16 + m) * H;
  v16h a0 = afrag(erow, 0, hi, 1.0f);
  v16h a1 = afrag(erow, 32, hi, 1.0f);
  float* tb = tbuf[wave];
#pragma unroll
  for (int nt = 0; nt < 2; ++nt) {
    v8f acc;
    float bv = b1[nt * 16 + m];
#pragma unroll
    for (int r = 0; r < 8; ++r) acc[r] = bv;
    acc = wmma16(a0, wf1[nt * 2 + 0], acc);
    acc = wmma16(a1, wf1[nt * 2 + 1], acc);
#pragma unroll
    for (int r = 0; r < 8; ++r)
      tb[(hi * 8 + r) * 36 + nt * 16 + m] = fmaxf(acc[r], 0.0f);
  }
  v16h a2 = afrag(tb + m * 36, 0, hi, 1.0f);
  v8f acc;
  float bv = b2[m];
#pragma unroll
  for (int r = 0; r < 8; ++r) acc[r] = bv;
  acc = wmma16(a2, wf2, acc);
#pragma unroll
  for (int r = 0; r < 8; ++r)
    q[((size_t)rb * 16 + hi * 8 + r) * AH + m] = acc[r];
}

// ---------------------------------------------------------------------------
static inline int cdiv(int a, int b) { return (a + b - 1) / b; }

extern "C" void kernel_launch(void* const* d_in, const int* in_sizes, int n_in,
                              void* d_out, int out_size, void* d_ws,
                              size_t ws_size, hipStream_t stream) {
  (void)in_sizes; (void)n_in; (void)out_size; (void)ws_size;

  const float* x_agv  = (const float*)d_in[0];
  const float* x_pick = (const float*)d_in[1];
  const float* x_loc  = (const float*)d_in[2];
  const int* e_src[6] = {(const int*)d_in[3],  (const int*)d_in[5],
                         (const int*)d_in[7],  (const int*)d_in[9],
                         (const int*)d_in[11], (const int*)d_in[13]};
  const int* e_dst[6] = {(const int*)d_in[4],  (const int*)d_in[6],
                         (const int*)d_in[8],  (const int*)d_in[10],
                         (const int*)d_in[12], (const int*)d_in[14]};
  const float* W_emb_agv  = (const float*)d_in[15];
  const float* b_emb_agv  = (const float*)d_in[16];
  const float* W_emb_pick = (const float*)d_in[17];
  const float* b_emb_pick = (const float*)d_in[18];
  const float* W_emb_loc  = (const float*)d_in[19];
  const float* b_emb_loc  = (const float*)d_in[20];
  const float* Wl     = (const float*)d_in[21];  // (2,6,64,64)
  const float* blv    = (const float*)d_in[22];  // (2,6,64)
  const float* Wr     = (const float*)d_in[23];  // (2,6,64,64)
  const float* W_ode1 = (const float*)d_in[24];
  const float* b_ode1 = (const float*)d_in[25];
  const float* W_ode2 = (const float*)d_in[26];
  const float* b_ode2 = (const float*)d_in[27];
  const float* W_ode3 = (const float*)d_in[28];
  const float* b_ode3 = (const float*)d_in[29];
  const float* W_ah1  = (const float*)d_in[30];
  const float* b_ah1  = (const float*)d_in[31];
  const float* W_ah2  = (const float*)d_in[32];
  const float* b_ah2  = (const float*)d_in[33];
  const float* W_ph1  = (const float*)d_in[34];
  const float* b_ph1  = (const float*)d_in[35];
  const float* W_ph2  = (const float*)d_in[36];
  const float* b_ph2  = (const float*)d_in[37];

  // d_out layout: agv_q | pick_q | agv_emb | pick_emb | loc_emb
  // The emb sections are contiguous -> use them directly as the ODE state X.
  float* out    = (float*)d_out;
  float* q_agv  = out;
  float* q_pick = out + (size_t)NA * AH;
  float* X      = out + (size_t)(NA + NPK) * AH;
  float* Xagv   = X;
  float* Xpick  = X + (size_t)NA * H;
  float* Xloc   = X + (size_t)(NA + NPK) * H;

  // Workspace: combine buffers, scatter accumulators, packed f16 weights.
  float* ws        = (float*)d_ws;
  float* comb_agv  = ws;
  float* comb_pick = comb_agv + (size_t)NA * H;
  float* comb_loc  = comb_pick + (size_t)NPK * H;
  float* sum       = comb_loc + (size_t)NL * H;
  float* cnt       = sum + (size_t)NL * H;
  _Float16* wl_pk  = (_Float16*)(cnt + NL);      // 32B-aligned by construction
  _Float16* wr_pk  = wl_pk + 12 * 4096;
  _Float16* w1_pk  = wr_pk + 12 * 4096;
  _Float16* w2_pk  = w1_pk + 2048;
  _Float16* w3_pk  = w2_pk + 1024;
  _Float16* ah1_pk = w3_pk + 2048;
  _Float16* ah2_pk = ah1_pk + 2048;
  _Float16* ph1_pk = ah2_pk + 512;
  _Float16* ph2_pk = ph1_pk + 2048;

  // ---- pack all weights into B-fragment f16 layout (tiny, L2-resident)
  pack_b<<<192, 256, 0, stream>>>(Wl, wl_pk, 64, 64, 12);
  pack_b<<<192, 256, 0, stream>>>(Wr, wr_pk, 64, 64, 12);
  pack_b<<<8, 256, 0, stream>>>(W_ode1, w1_pk, 64, 32, 1);
  pack_b<<<4, 256, 0, stream>>>(W_ode2, w2_pk, 32, 32, 1);
  pack_b<<<8, 256, 0, stream>>>(W_ode3, w3_pk, 32, 64, 1);
  pack_b<<<8, 256, 0, stream>>>(W_ah1, ah1_pk, 64, 32, 1);
  pack_b<<<2, 256, 0, stream>>>(W_ah2, ah2_pk, 32, 16, 1);
  pack_b<<<8, 256, 0, stream>>>(W_ph1, ph1_pk, 64, 32, 1);
  pack_b<<<2, 256, 0, stream>>>(W_ph2, ph2_pk, 32, 16, 1);

  // ---- embeddings straight into the state buffer
  embed<<<cdiv(NA * H, 256), 256, 0, stream>>>(x_agv, W_emb_agv, b_emb_agv, Xagv, NA, 7);
  embed<<<cdiv(NPK * H, 256), 256, 0, stream>>>(x_pick, W_emb_pick, b_emb_pick, Xpick, NPK, 4);
  embed<<<cdiv(NL * H, 256), 256, 0, stream>>>(x_loc, W_emb_loc, b_emb_loc, Xloc, NL, 2);

  // ---- hetero SAGE layers
  const float* hs[6] = {Xagv, Xloc, Xagv, Xpick, Xagv, Xpick};
  const float* hd[6] = {Xloc, Xagv, Xagv, Xloc, Xpick, Xagv};
  float*       cb[6] = {comb_loc, comb_agv, comb_agv, comb_loc, comb_pick, comb_agv};
  int          dn[6] = {NL, NA, NA, NL, NPK, NA};

  for (int l = 0; l < 2; ++l) {
    hipMemsetAsync(comb_agv, 0, (size_t)NA * H * 4, stream);
    hipMemsetAsync(comb_pick, 0, (size_t)NPK * H * 4, stream);
    hipMemsetAsync(comb_loc, 0, (size_t)NL * H * 4, stream);
    for (int s = 0; s < 6; ++s) {
      hipMemsetAsync(sum, 0, (size_t)dn[s] * H * 4, stream);
      hipMemsetAsync(cnt, 0, (size_t)dn[s] * 4, stream);
      edge_scatter<<<2048, 256, 0, stream>>>(hs[s], e_src[s], e_dst[s], sum, cnt, NE);
      int nrb = dn[s] / 16;
      sage_gemm<<<cdiv(nrb, SG_WAVES), SG_WAVES * 32, 0, stream>>>(
          sum, cnt, hd[s], wl_pk + (size_t)(l * 6 + s) * 4096,
          blv + (size_t)(l * 6 + s) * 64, wr_pk + (size_t)(l * 6 + s) * 4096,
          cb[s], nrb);
    }
    relu_scale<<<cdiv(NA * H, 256), 256, 0, stream>>>(comb_agv, Xagv, 1.0f / 3.0f, NA * H);
    relu_scale<<<cdiv(NL * H, 256), 256, 0, stream>>>(comb_loc, Xloc, 0.5f, NL * H);
    relu_scale<<<cdiv(NPK * H, 256), 256, 0, stream>>>(comb_pick, Xpick, 1.0f, NPK * H);
  }

  // ---- Neural-ODE RK4: 8 fully fused in-place steps over all 350k rows
  float dt = 1.0f / 8.0f;
  int nrb = NTOT / 16;
  for (int s = 0; s < 8; ++s)
    rk4_step<<<cdiv(nrb, RK_WAVES), RK_WAVES * 32, 0, stream>>>(
        X, w1_pk, b_ode1, w2_pk, b_ode2, w3_pk, b_ode3, nrb, dt);

  // ---- Q-value heads
  head_mlp<<<cdiv(NA / 16, HD_WAVES), HD_WAVES * 32, 0, stream>>>(
      Xagv, ah1_pk, b_ah1, ah2_pk, b_ah2, q_agv, NA / 16);
  head_mlp<<<cdiv(NPK / 16, HD_WAVES), HD_WAVES * 32, 0, stream>>>(
      Xpick, ph1_pk, b_ph1, ph2_pk, b_ph2, q_pick, NPK / 16);
}
